// RECT_L_64982855188849
// MI455X (gfx1250) — compile-verified, matrix operand order checked by reading, not run on hardware
//
#include <hip/hip_runtime.h>
#include <hip/hip_bf16.h>

// ---------------------------------------------------------------------------
// GCN layer on MI455X (gfx1250, wave32):
//   deg/dinv -> h = x@W_conv (WMMA bf16) -> agg = b_conv + dinv^2*h + scatter
//   -> out = agg@W_lin + b_lin (WMMA bf16)
// GEMMs: double-buffered LDS pipeline, one barrier per 32-K step, uniform
// (divergence-free) staging, global loads for step s+1 in flight while
// v_wmma_f32_16x16x32_bf16 consumes step s from LDS.
// ---------------------------------------------------------------------------

typedef __attribute__((ext_vector_type(16))) __bf16 v16bf;
typedef __attribute__((ext_vector_type(8)))  __bf16 v8bf;
typedef __attribute__((ext_vector_type(2)))  __bf16 v2bf;
typedef __attribute__((ext_vector_type(8)))  float  v8f;

#define NODES 50000
#define EDGES 1600000
#define FIN   512
#define HID   128

// ---------------- degree / norm kernels ----------------

__global__ void deg_init_kernel(float* __restrict__ deg, int n) {
    int i = blockIdx.x * blockDim.x + threadIdx.x;
    if (i < n) deg[i] = 1.0f;   // self-loop weight
}

__global__ void deg_edges_kernel(const int* __restrict__ dst,
                                 const float* __restrict__ w,
                                 float* __restrict__ deg, int e) {
    int i = blockIdx.x * blockDim.x + threadIdx.x;
    if (i < e) unsafeAtomicAdd(&deg[dst[i]], w[i]);
}

__global__ void dinv_kernel(float* __restrict__ deg, int n) {
    int i = blockIdx.x * blockDim.x + threadIdx.x;
    if (i < n) {
        float d = deg[i];
        deg[i] = (d > 0.0f) ? rsqrtf(d) : 0.0f;
    }
}

// ---------------- weight pre-transpose: Wt[c][k] = (bf16)W[k][c] ------------

__global__ void wtrans_kernel(const float* __restrict__ W,
                              __bf16* __restrict__ Wt, int K, int Ntot) {
    int gid = blockIdx.x * blockDim.x + threadIdx.x;
    if (gid < K * Ntot) {
        int c = gid / K;
        int k = gid - c * K;
        Wt[gid] = (__bf16)W[(size_t)k * Ntot + c];
    }
}

// ---------------- WMMA GEMM: C[M x NTOT] = A[M x K] @ W[K x NTOT] (+bias) ---
// Block = 256 threads = 8 wave32; covers 16 rows x 128 cols
// (blockIdx.x -> 16-row tile, blockIdx.y -> 128-col block). Each wave owns a
// 16x16 C tile. Double-buffered LDS, one s_barrier per step. LDS rows padded
// to 40 elems (80B stride: 16B-aligned b128, bank-shifted across rows).

template <int K, int NTOT>
__global__ void wmma_gemm16_kernel(const float*  __restrict__ A,
                                   const __bf16* __restrict__ Wt,   // [NTOT][K]
                                   float*        __restrict__ C,
                                   const float*  __restrict__ bias) {
    constexpr int STEPS = K / 32;
    __shared__ __align__(16) __bf16 atile[2][16][40];   // 16 x 32 (+8 pad)
    __shared__ __align__(16) __bf16 wtile[2][128][40];  // 128 x 32 (+8 pad)

    const int tid  = threadIdx.x;
    const int lane = tid & 31;
    const int wave = tid >> 5;
    const int r0   = blockIdx.x * 16;
    const int cb   = blockIdx.y * 128;

    const int r   = lane & 15;               // row (A) / col-in-tile (B,C)
    const int klo = (lane < 16) ? 0 : 8;     // ISA 16-bit A/B lane K split
    const int c   = wave * 16 + r;           // column within 128-col block

    // Uniform staging coordinates (all 256 threads participate).
    const int arow = tid >> 4;               // A row 0..15
    const int ak2  = (tid & 15) << 1;        // A k-pair (2 floats)
    const int cc0  = tid >> 2;               // W col 0..63
    const int q8   = (tid & 3) << 3;         // W k-octet
    const int cc1  = cc0 + 64;               // W col 64..127

    const float*  aBase  = A  + (size_t)(r0 + arow) * K + ak2;
    const __bf16* wBase0 = Wt + (size_t)(cb + cc0) * K + q8;
    const __bf16* wBase1 = Wt + (size_t)(cb + cc1) * K + q8;

    float2 a2;                               // in-flight staged data
    v8bf   wv0, wv1;

    auto loadStage = [&](int s) {            // global loads for step s
        const int k0 = s * 32;
        a2  = *(const float2*)(aBase + k0);
        wv0 = *(const v8bf*)(wBase0 + k0);
        wv1 = *(const v8bf*)(wBase1 + k0);
    };
    auto storeStage = [&](int s) {           // deposit into LDS buffer s&1
        const int b = s & 1;
        v2bf pk = { (__bf16)a2.x, (__bf16)a2.y };   // one v_cvt_pk_bf16_f32
        *(v2bf*)&atile[b][arow][ak2] = pk;          // ds_store_b32
        *(v8bf*)&wtile[b][cc0][q8] = wv0;           // ds_store_b128 x2
        *(v8bf*)&wtile[b][cc1][q8] = wv1;
    };

    v8f acc = {};

    loadStage(0);
    storeStage(0);

#pragma unroll
    for (int s = 0; s < STEPS; ++s) {
        __syncthreads();                     // buffer s&1 ready for reads
        if (s + 1 < STEPS) loadStage(s + 1); // loads overlap the WMMA below

        const int b = s & 1;
        // A fragment (16x32 bf16): lanes 0-15 -> K {0..7,16..23},
        //                          lanes 16-31 -> K {8..15,24..31}
        v8bf alo = *(const v8bf*)&atile[b][r][klo];
        v8bf ahi = *(const v8bf*)&atile[b][r][klo + 16];
        v16bf avf = __builtin_shufflevector(alo, ahi,
            0, 1, 2, 3, 4, 5, 6, 7, 8, 9, 10, 11, 12, 13, 14, 15);
        // B fragment (32x16 bf16), mirrored layout from transposed slab.
        v8bf blo = *(const v8bf*)&wtile[b][c][klo];
        v8bf bhi = *(const v8bf*)&wtile[b][c][klo + 16];
        v16bf bvf = __builtin_shufflevector(blo, bhi,
            0, 1, 2, 3, 4, 5, 6, 7, 8, 9, 10, 11, 12, 13, 14, 15);

        acc = __builtin_amdgcn_wmma_f32_16x16x32_bf16(
            /*neg_a=*/false, avf, /*neg_b=*/false, bvf,
            /*c_mod=*/(short)0, acc, /*reuse_a=*/false, /*reuse_b=*/false);

        if (s + 1 < STEPS) storeStage(s + 1);  // opposite buffer; next barrier
    }                                           // publishes it

    // Epilogue: f32 C/D layout -> VGPR v holds row (v + 8*(lane>=16)), col r.
    const int col = cb + wave * 16 + r;
    const float badd = bias ? bias[col] : 0.0f;
    const int mbase = (lane < 16) ? 0 : 8;
#pragma unroll
    for (int v = 0; v < 8; ++v) {
        C[(size_t)(r0 + v + mbase) * NTOT + col] = acc[v] + badd;
    }
}

// ---------------- aggregation kernels ----------------

// agg[i][c] = b_conv[c] + dinv[i]^2 * h[i][c]   (self-loop + bias)
__global__ void agg_init_kernel(const float* __restrict__ h,
                                const float* __restrict__ dinv,
                                const float* __restrict__ bconv,
                                float* __restrict__ agg, long total) {
    long gid = (long)blockIdx.x * blockDim.x + threadIdx.x;
    if (gid < total) {
        int i  = (int)(gid >> 7);      // HID = 128
        int cc = (int)(gid & 127);
        float di = dinv[i];
        agg[gid] = bconv[cc] + di * di * h[gid];
    }
}

// 32 threads per edge; each thread: float4 of h[src] (L2-resident), 4 native
// f32 atomic adds into agg[dst] (25.6MB << 192MB L2).
__global__ void edge_agg_kernel(const int* __restrict__ src,
                                const int* __restrict__ dst,
                                const float* __restrict__ w,
                                const float* __restrict__ dinv,
                                const float* __restrict__ h,
                                float* __restrict__ agg, int e) {
    long gid = (long)blockIdx.x * blockDim.x + threadIdx.x;
    int ei = (int)(gid >> 5);
    if (ei >= e) return;
    int cg = (int)(gid & 31) << 2;     // channel group of 4
    int s = src[ei], d = dst[ei];
    float nm = dinv[s] * w[ei] * dinv[d];
    const float4 hv = *(const float4*)&h[(size_t)s * HID + cg];
    float* ap = &agg[(size_t)d * HID + cg];
    unsafeAtomicAdd(ap + 0, hv.x * nm);
    unsafeAtomicAdd(ap + 1, hv.y * nm);
    unsafeAtomicAdd(ap + 2, hv.z * nm);
    unsafeAtomicAdd(ap + 3, hv.w * nm);
}

// ---------------- launch ----------------

extern "C" void kernel_launch(void* const* d_in, const int* in_sizes, int n_in,
                              void* d_out, int out_size, void* d_ws, size_t ws_size,
                              hipStream_t stream) {
    const float* x      = (const float*)d_in[0];        // [N, FIN]
    const int*   eidx   = (const int*)  d_in[1];        // [2, E]
    const float* ew     = (const float*)d_in[2];        // [E]
    const float* W_conv = (const float*)d_in[3];        // [FIN, HID]
    const float* b_conv = (const float*)d_in[4];        // [HID]
    const float* W_lin  = (const float*)d_in[5];        // [HID, FIN]
    const float* b_lin  = (const float*)d_in[6];        // [FIN]
    float*       out    = (float*)d_out;                // [N, FIN]

    const int* src = eidx;
    const int* dst = eidx + EDGES;

    // Workspace: dinv[N] | h[N*HID] | agg[N*HID] (f32) | WconvT | WlinT (bf16)
    float* dinv = (float*)d_ws;
    float* h    = dinv + NODES;
    float* agg  = h + (size_t)NODES * HID;
    __bf16* WconvT = (__bf16*)(agg + (size_t)NODES * HID);   // [HID][FIN]
    __bf16* WlinT  = WconvT + (size_t)FIN * HID;             // [FIN][HID]

    const int T = 256;

    // 1) degree with self-loops, then dinv = rsqrt(deg)
    deg_init_kernel<<<(NODES + T - 1) / T, T, 0, stream>>>(dinv, NODES);
    deg_edges_kernel<<<(EDGES + T - 1) / T, T, 0, stream>>>(dst, ew, dinv, EDGES);
    dinv_kernel<<<(NODES + T - 1) / T, T, 0, stream>>>(dinv, NODES);

    // 2) pre-transpose weights to bf16 [col][k]
    wtrans_kernel<<<(FIN * HID + T - 1) / T, T, 0, stream>>>(W_conv, WconvT, FIN, HID);
    wtrans_kernel<<<(FIN * HID + T - 1) / T, T, 0, stream>>>(W_lin,  WlinT,  HID, FIN);

    // 3) h = x @ W_conv   (50000x512 @ 512x128)
    wmma_gemm16_kernel<FIN, HID><<<dim3(NODES / 16, 1), T, 0, stream>>>(
        x, WconvT, h, nullptr);

    // 4) agg = b_conv + dinv^2 * h, then scatter-add over edges
    {
        long total = (long)NODES * HID;
        agg_init_kernel<<<(unsigned)((total + T - 1) / T), T, 0, stream>>>(
            h, dinv, b_conv, agg, total);
    }
    {
        long total = (long)EDGES * 32;
        edge_agg_kernel<<<(unsigned)((total + T - 1) / T), T, 0, stream>>>(
            src, dst, ew, dinv, h, agg, EDGES);
    }

    // 5) out = agg @ W_lin + b_lin  (50000x128 @ 128x512)
    wmma_gemm16_kernel<HID, FIN><<<dim3(NODES / 16, FIN / 128), T, 0, stream>>>(
        agg, WlinT, out, b_lin);
}
